// PhpNetGraphTokensCombineFileLevel_62010737820210
// MI455X (gfx1250) — matrix operand order repaired and checked by reading, not compiled, not run on hardware
//
#include <hip/hip_runtime.h>

// ---------------------------------------------------------------------------
// Types / WMMA fragment helpers (gfx1250, wave32)
// ---------------------------------------------------------------------------
typedef __attribute__((ext_vector_type(16))) __bf16 bf16x16;
typedef __attribute__((ext_vector_type(8)))  float  f32x8;

#define TM 128
#define TN 128
#define KBL 32
#define KB2 16  // KBL/2 packed bf16 pairs

__device__ __forceinline__ unsigned pack_bf16(float a, float b) {
  unsigned ua = __float_as_uint(a), ub = __float_as_uint(b);
  unsigned ra = (ua + 0x7FFFu + ((ua >> 16) & 1u)) >> 16;  // RNE
  unsigned rb = (ub + 0x7FFFu + ((ub >> 16) & 1u)) >> 16;
  return (ra & 0xFFFFu) | (rb << 16);
}
__device__ __forceinline__ unsigned short bf16r(float a) {
  unsigned u = __float_as_uint(a);
  return (unsigned short)((u + 0x7FFFu + ((u >> 16) & 1u)) >> 16);
}

union FragAB { unsigned u[8]; bf16x16 v; };

// ---------------------------------------------------------------------------
// Operand pre-conversion kernels (fp32 -> packed bf16)
// ---------------------------------------------------------------------------
__global__ void conv_bf16(const float* __restrict__ in, unsigned* __restrict__ out, long npairs) {
  long i = (long)blockIdx.x * blockDim.x + threadIdx.x;
  if (i >= npairs) return;
  out[i] = pack_bf16(in[2 * i], in[2 * i + 1]);
}
// W[K,N] fp32 -> Wt[N,K] bf16 (LDS-tiled transpose)
__global__ void conv_transpose_bf16(const float* __restrict__ W, unsigned short* __restrict__ Wt,
                                    int K, int N) {
  __shared__ float tile[32][33];
  int k0 = blockIdx.y * 32, n0 = blockIdx.x * 32;
  int tx = threadIdx.x & 31, ty = threadIdx.x >> 5;  // 32x8
  for (int i = ty; i < 32; i += 8) {
    int k = k0 + i, n = n0 + tx;
    tile[i][tx] = (k < K && n < N) ? W[(long)k * N + n] : 0.f;
  }
  __syncthreads();
  for (int i = ty; i < 32; i += 8) {
    int n = n0 + i, k = k0 + tx;
    if (n < N && k < K) Wt[(long)n * K + k] = bf16r(tile[tx][i]);
  }
}

// ---------------------------------------------------------------------------
// bf16 WMMA GEMM: C[M,N] = A[M,K] * B (+bias)(+relu)
//   A16: packed bf16 [M,K] (lda elements, even); B16: packed bf16 [N,K] (ldb)
//   block tile 128x128, 8 waves; wave tile 32x64 (2x4 wmma accumulators)
//   double-buffered LDS staging, prefetch on the A stream
// ---------------------------------------------------------------------------
__global__ __launch_bounds__(256) void gemm_bf16_wmma(
    const unsigned* __restrict__ A16, const unsigned* __restrict__ B16,
    const float* __restrict__ bias, float* __restrict__ C,
    int M, int N, int K, int lda, int ldb, int ldc, int relu)
{
  __shared__ unsigned Asm[2][TM * KB2];  // [buf][m][kpair]
  __shared__ unsigned Bsm[2][TN * KB2];  // [buf][n][kpair]
  const int tid = threadIdx.x;
  const int rowBase = blockIdx.y * TM;
  const int colBase = blockIdx.x * TN;
  const int lane = tid & 31;
  const int wave = tid >> 5;
  const int wM = wave & 3;     // 4 M slots of 32 rows
  const int wN = wave >> 2;    // 2 N slots of 64 cols
  const int l15 = lane & 15;
  const int hi = lane >> 4;

  f32x8 acc[2][4] = {};

  auto stage = [&](int buf, int kb) {
    unsigned* As = Asm[buf];
    unsigned* Bs = Bsm[buf];
    for (int p = tid; p < TM * KB2; p += 256) {
      int kp = p & (KB2 - 1);
      int mr = p >> 4;
      int row = rowBase + mr;
      int k0 = kb + 2 * kp;
      As[p] = (row < M && k0 < K) ? A16[((long)row * lda + k0) >> 1] : 0u;
      if (kp == 0 && row < M && kb + 2 * KBL < K)
        __builtin_prefetch(&A16[((long)row * lda + kb + 2 * KBL) >> 1], 0, 0);
    }
    for (int p = tid; p < TN * KB2; p += 256) {
      int kp = p & (KB2 - 1);
      int nc = p >> 4;
      int col = colBase + nc;
      int k0 = kb + 2 * kp;
      Bs[p] = (col < N && k0 < K) ? B16[((long)col * ldb + k0) >> 1] : 0u;
    }
  };

  const int nk = (K + KBL - 1) / KBL;
  stage(0, 0);
  __syncthreads();

  for (int ib = 0; ib < nk; ++ib) {
    const int cur = ib & 1;
    if (ib + 1 < nk) stage(cur ^ 1, (ib + 1) * KBL);

    // A 16-bit 16x32 layout: lane<16 holds K {0..7,16..23}, lane>=16 {8..15,24..31}
    FragAB af[2];
#pragma unroll
    for (int mt = 0; mt < 2; ++mt) {
      const unsigned* base = &Asm[cur][(wM * 32 + mt * 16 + l15) * KB2];
#pragma unroll
      for (int v = 0; v < 8; ++v)
        af[mt].u[v] = base[((v < 4) ? v : v + 4) + (hi ? 4 : 0)];
    }
    // B 32x16: lane<16 col=l15 K {0..7,16..23}; lane>=16 col=l15 K {8..15,24..31}
    FragAB bfr[4];
#pragma unroll
    for (int nt = 0; nt < 4; ++nt) {
      const unsigned* base = &Bsm[cur][(wN * 64 + nt * 16 + l15) * KB2];
#pragma unroll
      for (int v = 0; v < 8; ++v)
        bfr[nt].u[v] = base[((v < 4) ? v : v + 4) + (hi ? 4 : 0)];
    }
#pragma unroll
    for (int mt = 0; mt < 2; ++mt)
#pragma unroll
      for (int nt = 0; nt < 4; ++nt)
        acc[mt][nt] = __builtin_amdgcn_wmma_f32_16x16x32_bf16(
            false, af[mt].v, false, bfr[nt].v, (short)0, acc[mt][nt], false, false);
    __syncthreads();
  }

  // C/D layout: VGPR r -> M = r + (hi?8:0); N = l15
#pragma unroll
  for (int mt = 0; mt < 2; ++mt) {
#pragma unroll
    for (int nt = 0; nt < 4; ++nt) {
      int col = colBase + wN * 64 + nt * 16 + l15;
#pragma unroll
      for (int r = 0; r < 8; ++r) {
        int row = rowBase + wM * 32 + mt * 16 + (hi ? 8 : 0) + r;
        if (row < M && col < N) {
          float v = acc[mt][nt][r];
          if (bias) v += bias[col];
          if (relu) v = fmaxf(v, 0.f);
          C[(long)row * ldc + col] = v;
        }
      }
    }
  }
}

// ---------------------------------------------------------------------------
// Utility kernels
// ---------------------------------------------------------------------------
__global__ void fill_f(float* p, float v, long n) {
  long i = (long)blockIdx.x * blockDim.x + threadIdx.x;
  if (i < n) p[i] = v;
}
__global__ void fill_i(int* p, int v, long n) {
  long i = (long)blockIdx.x * blockDim.x + threadIdx.x;
  if (i < n) p[i] = v;
}
__global__ void embed_gather(const int* __restrict__ idx, const float* __restrict__ tab,
                             float* __restrict__ out, long rows, int D) {
  long i = (long)blockIdx.x * blockDim.x + threadIdx.x;
  if (i >= rows * (long)D) return;
  long r = i / D; int e = (int)(i % D);
  out[i] = tab[(long)idx[r] * D + e];
}

__device__ __forceinline__ void atomicMaxF(float* addr, float val) {
  unsigned* ua = (unsigned*)addr;
  unsigned old = *ua;
  while (__uint_as_float(old) < val) {
    unsigned assumed = old;
    old = atomicCAS(ua, assumed, __float_as_uint(val));
    if (old == assumed) break;
  }
}

// ---------------------------------------------------------------------------
// GCN pieces
// ---------------------------------------------------------------------------
__global__ void deg_edges(const int* __restrict__ ei, int E, float* deg) {
  int e = blockIdx.x * blockDim.x + threadIdx.x;
  if (e >= E) return;
  atomicAdd(&deg[ei[E + e]], 1.0f);
}
__global__ void rsqrt_k(const float* deg, float* dinv, int N) {
  int i = blockIdx.x * blockDim.x + threadIdx.x;
  if (i < N) dinv[i] = rsqrtf(deg[i]);
}
__global__ void gcn_self(const float* __restrict__ h, const float* __restrict__ dinv,
                         const float* __restrict__ bias, float* __restrict__ out,
                         int N, int C) {
  long i = (long)blockIdx.x * blockDim.x + threadIdx.x;
  if (i >= (long)N * C) return;
  int n = (int)(i / C); int c = (int)(i % C);
  float di = dinv[n];
  out[i] = h[i] * di * di + bias[c];
}
__global__ void gcn_edges(const int* __restrict__ ei, int E, const float* __restrict__ h,
                          const float* __restrict__ dinv, float* __restrict__ out, int C) {
  int e = blockIdx.x;
  int s = ei[e], d = ei[E + e];
  float nm = dinv[s] * dinv[d];
  for (int c = threadIdx.x; c < C; c += blockDim.x)
    atomicAdd(&out[(long)d * C + c], h[(long)s * C + c] * nm);
}

// ---------------------------------------------------------------------------
// EdgePooling pieces
// ---------------------------------------------------------------------------
__global__ void pool_dots(const float* __restrict__ x, const float* __restrict__ w, int C,
                          float* __restrict__ ys, float* __restrict__ yd) {
  int n = blockIdx.x;
  __shared__ float s1[256], s2[256];
  float a = 0.f, b = 0.f;
  for (int c = threadIdx.x; c < C; c += 256) {
    float xv = x[(long)n * C + c];
    a += xv * w[c];
    b += xv * w[C + c];
  }
  s1[threadIdx.x] = a; s2[threadIdx.x] = b;
  __syncthreads();
  for (int st = 128; st > 0; st >>= 1) {
    if (threadIdx.x < st) {
      s1[threadIdx.x] += s1[threadIdx.x + st];
      s2[threadIdx.x] += s2[threadIdx.x + st];
    }
    __syncthreads();
  }
  if (threadIdx.x == 0) { ys[n] = s1[0]; yd[n] = s2[0]; }
}
__global__ void edge_raw(const int* __restrict__ ei, int E, const float* ys, const float* yd,
                         const float* bsc, float* raw, float* m) {
  int e = blockIdx.x * blockDim.x + threadIdx.x;
  if (e >= E) return;
  int s = ei[e], d = ei[E + e];
  float r = ys[s] + yd[d] + bsc[0];
  raw[e] = r;
  atomicMaxF(&m[d], r);
}
__global__ void edge_exp(const int* __restrict__ ei, int E, const float* raw, const float* m,
                         float* exv, float* den) {
  int e = blockIdx.x * blockDim.x + threadIdx.x;
  if (e >= E) return;
  int d = ei[E + e];
  float ex = expf(raw[e] - m[d]);
  exv[e] = ex;
  atomicAdd(&den[d], ex);
}
__global__ void edge_key(const int* __restrict__ ei, int E, int PAD, const float* exv,
                         const float* den, float* score, unsigned long long* keys) {
  int e = blockIdx.x * blockDim.x + threadIdx.x;
  if (e >= PAD) return;
  if (e < E) {
    int d = ei[E + e];
    float sc = exv[e] / den[d] + 0.5f;
    score[e] = sc;
    unsigned u = __float_as_uint(sc);
    u = (u & 0x80000000u) ? ~u : (u | 0x80000000u);            // orderable-ascending bits
    keys[e] = ((unsigned long long)(~u) << 32) | (unsigned)e;  // desc score, asc idx (stable)
  } else {
    keys[e] = 0xFFFFFFFFFFFFFFFFULL;
  }
}
__global__ void bitonic_step(unsigned long long* k, int j, int kk) {
  int i = blockIdx.x * blockDim.x + threadIdx.x;
  int ixj = i ^ j;
  if (ixj > i) {
    unsigned long long a = k[i], b = k[ixj];
    bool up = ((i & kk) == 0);
    if ((up && a > b) || (!up && a < b)) { k[i] = b; k[ixj] = a; }
  }
}
__global__ void pool_init(int* merged, int* cluster, float* nscore, int N) {
  int i = blockIdx.x * blockDim.x + threadIdx.x;
  if (i >= N) return;
  merged[i] = 0; cluster[i] = i; nscore[i] = 1.0f;
}
__global__ void pool_scan(const unsigned long long* __restrict__ keys,
                          const int* __restrict__ ei, int E,
                          const float* __restrict__ score,
                          int* merged, int* cluster, float* nscore, int* valid) {
  if (threadIdx.x != 0 || blockIdx.x != 0) return;  // strictly sequential greedy merge
  for (int i = 0; i < E; ++i) {
    int e = (int)(keys[i] & 0xFFFFFFFFu);
    int s = ei[e], d = ei[E + e];
    if (!merged[s] && !merged[d] && s != d) {
      int rep = s < d ? s : d;
      int oth = s < d ? d : s;
      merged[s] = 1; merged[d] = 1;
      cluster[oth] = rep;
      nscore[rep] = score[e];
      valid[oth] = 0;
    }
  }
}
__global__ void pool_agg(const float* __restrict__ x, const int* __restrict__ cluster,
                         float* __restrict__ nx, int C) {
  int n = blockIdx.x;
  int cl = cluster[n];
  for (int c = threadIdx.x; c < C; c += blockDim.x)
    atomicAdd(&nx[(long)cl * C + c], x[(long)n * C + c]);
}
__global__ void pool_scale_relu(float* __restrict__ nx, const float* __restrict__ nscore,
                                int N, int C) {
  long i = (long)blockIdx.x * blockDim.x + threadIdx.x;
  if (i >= (long)N * C) return;
  int n = (int)(i / C);
  nx[i] = fmaxf(nx[i] * nscore[n], 0.f);
}
__global__ void remap_edges(const int* __restrict__ ei, const int* __restrict__ cluster,
                            int* __restrict__ eo, int n2e) {
  int i = blockIdx.x * blockDim.x + threadIdx.x;
  if (i < n2e) eo[i] = cluster[ei[i]];
}

// ---------------------------------------------------------------------------
// Global max pool
// ---------------------------------------------------------------------------
__global__ void batch_bounds(const int* __restrict__ batch, int N, int B, int* start) {
  int b = threadIdx.x;
  if (b > B) return;
  int lo = 0, hi = N;
  while (lo < hi) { int mid = (lo + hi) >> 1; if (batch[mid] < b) lo = mid + 1; else hi = mid; }
  start[b] = lo;
}
__global__ void gmax(const float* __restrict__ x, const int* __restrict__ valid,
                     const int* __restrict__ start, int C, float* __restrict__ xg) {
  int b = blockIdx.y;
  int c = blockIdx.x * blockDim.x + threadIdx.x;
  if (c >= C) return;
  float m = -1e30f;
  int n0 = start[b], n1 = start[b + 1];
  for (int n = n0; n < n1; ++n) {
    float v = valid[n] ? x[(long)n * C + c] : -1e30f;
    m = fmaxf(m, v);
  }
  xg[(long)b * C + c] = m;
}

// ---------------------------------------------------------------------------
// GRU recurrence (persistent workgroup per direction); input projections done
// by WMMA GEMM beforehand: GI[b*T+t, 600] = x_t @ W_ih^T + b_ih
// ---------------------------------------------------------------------------
__global__ __launch_bounds__(1024) void gru_rec(
    const float* __restrict__ GIf, const float* __restrict__ GIb,
    const float* __restrict__ Whh,  // [2,600,200]
    const float* __restrict__ bhh,  // [2,600]
    float* __restrict__ Y,          // [B,T,400]
    float* __restrict__ x1,         // [B,1200]
    int layer, int T)
{
  const int dir = blockIdx.x;
  const float* GI = dir ? GIb : GIf;
  const float* W  = Whh + (long)dir * 600 * 200;
  const float* bh = bhh + dir * 600;
  __shared__ float hs[16 * 200];
  __shared__ float gh[16 * 600];
  const int tid = threadIdx.x;
  for (int i = tid; i < 16 * 200; i += blockDim.x) hs[i] = 0.f;
  __syncthreads();
  for (int t = 0; t < T; ++t) {
    int tau = dir ? (T - 1 - t) : t;
    for (int i = tid; i < 16 * 600; i += blockDim.x) {
      int b = i / 600, j = i % 600;
      const float* w = W + (long)j * 200;
      const float* h = hs + b * 200;
      float acc = bh[j];
      for (int k = 0; k < 200; ++k) acc += h[k] * w[k];
      gh[i] = acc;
    }
    __syncthreads();
    for (int i = tid; i < 16 * 200; i += blockDim.x) {
      int b = i / 200, hh = i % 200;
      const float* gi = GI + ((long)b * T + tau) * 600;
      float ir = gi[hh], iz = gi[200 + hh], in = gi[400 + hh];
      float hr = gh[b * 600 + hh], hz = gh[b * 600 + 200 + hh], hn = gh[b * 600 + 400 + hh];
      float r = 1.f / (1.f + expf(-(ir + hr)));
      float z = 1.f / (1.f + expf(-(iz + hz)));
      float n = tanhf(in + r * hn);
      float h = (1.f - z) * n + z * hs[i];
      hs[i] = h;
      Y[((long)b * T + tau) * 400 + dir * 200 + hh] = h;
    }
    __syncthreads();
  }
  for (int i = tid; i < 16 * 200; i += blockDim.x) {
    int b = i / 200, hh = i % 200;
    x1[b * 1200 + (2 * layer + dir) * 200 + hh] = hs[i];
  }
}

__global__ void concat_z(const float* __restrict__ xg, const float* __restrict__ x1,
                         float* __restrict__ z) {
  int i = blockIdx.x * blockDim.x + threadIdx.x;
  if (i >= 16 * 5200) return;
  int b = i / 5200, c = i % 5200;
  z[i] = (c < 4000) ? xg[b * 4000 + c] : x1[b * 1200 + (c - 4000)];
}

// ---------------------------------------------------------------------------
// Host side
// ---------------------------------------------------------------------------
static inline dim3 g1(long n, int b = 256) { return dim3((unsigned)((n + b - 1) / b)); }

static inline void gemm16(const unsigned* A16, const unsigned* B16, const float* bias, float* C,
                          int M, int N, int K, int lda, int ldb, int ldc,
                          int relu, hipStream_t s) {
  dim3 g((N + TN - 1) / TN, (M + TM - 1) / TM);
  gemm_bf16_wmma<<<g, 256, 0, s>>>(A16, B16, bias, C, M, N, K, lda, ldb, ldc, relu);
}

extern "C" void kernel_launch(void* const* d_in, const int* in_sizes, int n_in,
                              void* d_out, int out_size, void* d_ws, size_t ws_size,
                              hipStream_t stream) {
  const int* graph_x   = (const int*)d_in[0];
  const int* edge_idx  = (const int*)d_in[1];
  const int* batch     = (const int*)d_in[2];
  const int* tokens    = (const int*)d_in[3];
  const float* embed1w = (const float*)d_in[4];
  const float* embedw  = (const float*)d_in[5];
  const float* gcnW[3] = {(const float*)d_in[6], (const float*)d_in[8], (const float*)d_in[10]};
  const float* gcnB[3] = {(const float*)d_in[7], (const float*)d_in[9], (const float*)d_in[11]};
  const float* poolW[3] = {(const float*)d_in[12], (const float*)d_in[14], (const float*)d_in[16]};
  const float* poolB[3] = {(const float*)d_in[13], (const float*)d_in[15], (const float*)d_in[17]};
  const float* lin1w = (const float*)d_in[30]; const float* lin1b = (const float*)d_in[31];
  const float* lin2w = (const float*)d_in[32]; const float* lin2b = (const float*)d_in[33];
  const float* lin3w = (const float*)d_in[34]; const float* lin3b = (const float*)d_in[35];

  const int N = in_sizes[2];        // 10000 nodes
  const int E = in_sizes[1] / 2;    // 40000 edges
  const int NB = 16;                // graphs / batch
  const int T = in_sizes[3] / NB;   // 256 tokens
  const int ED = 100, H = 200;
  int PAD = 1; while (PAD < E) PAD <<= 1;

  // workspace carve
  char* wp = (char*)d_ws; size_t off = 0;
  auto alloc = [&](size_t bytes) -> void* {
    void* p = wp + off; off = (off + bytes + 255) & ~(size_t)255; return p;
  };
  float* X   = (float*)alloc((size_t)N * 4000 * 4);
  float* Hb  = (float*)alloc((size_t)N * 4000 * 4);
  float* T1  = (float*)alloc((size_t)N * 4000 * 4);
  unsigned* A16 = (unsigned*)alloc((size_t)N * 4000 * 2);        // packed bf16 A operand
  unsigned* Wt  = (unsigned*)alloc((size_t)4000 * 4000 * 2);     // packed bf16 [N,K] weights
  float* deg = (float*)alloc(N * 4); float* dinv = (float*)alloc(N * 4);
  float* ys  = (float*)alloc(N * 4); float* yd  = (float*)alloc(N * 4);
  float* mv  = (float*)alloc(N * 4); float* den = (float*)alloc(N * 4);
  float* rawv = (float*)alloc(E * 4); float* exv = (float*)alloc(E * 4);
  float* scv  = (float*)alloc(E * 4);
  unsigned long long* keys = (unsigned long long*)alloc((size_t)PAD * 8);
  int* merged = (int*)alloc(N * 4); int* cluster = (int*)alloc(N * 4);
  float* nscore = (float*)alloc(N * 4); int* valid = (int*)alloc(N * 4);
  int* eiA = (int*)alloc((size_t)2 * E * 4); int* eiB = (int*)alloc((size_t)2 * E * 4);
  int* startb = (int*)alloc((NB + 1) * 4);
  float* xg  = (float*)alloc((size_t)NB * 4000 * 4);
  float* tokemb = (float*)alloc((size_t)NB * T * ED * 4);
  float* Y0  = (float*)alloc((size_t)NB * T * 2 * H * 4);
  float* Y1  = (float*)alloc((size_t)NB * T * 2 * H * 4);
  float* GIf = (float*)alloc((size_t)NB * T * 3 * H * 4);
  float* GIb = (float*)alloc((size_t)NB * T * 3 * H * 4);
  float* x1  = (float*)alloc((size_t)NB * 6 * H * 4);
  float* z   = (float*)alloc((size_t)NB * 5200 * 4);
  float* z1  = (float*)alloc((size_t)NB * 1000 * 4);
  float* z2  = (float*)alloc((size_t)NB * 500 * 4);
  (void)ws_size; (void)n_in; (void)out_size;

  auto conv = [&](const float* src, unsigned* dst, long nElem) {  // nElem even
    conv_bf16<<<g1(nElem / 2), 256, 0, stream>>>(src, dst, nElem / 2);
  };
  auto convT = [&](const float* W, unsigned* WtB, int K, int Ncols) {
    conv_transpose_bf16<<<dim3((Ncols + 31) / 32, (K + 31) / 32), 256, 0, stream>>>(
        W, (unsigned short*)WtB, K, Ncols);
  };

  auto run_gcn = [&](int Cin, int Cout, const float* W, const float* bias, const int* ei) {
    fill_f<<<g1(N), 256, 0, stream>>>(deg, 1.0f, N);  // self loops
    deg_edges<<<g1(E), 256, 0, stream>>>(ei, E, deg);
    rsqrt_k<<<g1(N), 256, 0, stream>>>(deg, dinv, N);
    conv(X, A16, (long)N * Cin);
    convT(W, Wt, Cin, Cout);
    gemm16(A16, Wt, nullptr, Hb, N, Cout, Cin, Cin, Cin, Cout, 0, stream);  // h = x @ W
    gcn_self<<<g1((long)N * Cout), 256, 0, stream>>>(Hb, dinv, bias, T1, N, Cout);
    gcn_edges<<<dim3(E), 256, 0, stream>>>(ei, E, Hb, dinv, T1, Cout);
  };
  auto run_pool = [&](int C, const float* w, const float* bsc, const int* ei, int* eo) {
    pool_dots<<<dim3(N), 256, 0, stream>>>(T1, w, C, ys, yd);
    fill_f<<<g1(N), 256, 0, stream>>>(mv, -3.0e38f, N);
    fill_f<<<g1(N), 256, 0, stream>>>(den, 0.0f, N);
    edge_raw<<<g1(E), 256, 0, stream>>>(ei, E, ys, yd, bsc, rawv, mv);
    edge_exp<<<g1(E), 256, 0, stream>>>(ei, E, rawv, mv, exv, den);
    edge_key<<<g1(PAD), 256, 0, stream>>>(ei, E, PAD, exv, den, scv, keys);
    for (int kk = 2; kk <= PAD; kk <<= 1)
      for (int j = kk >> 1; j > 0; j >>= 1)
        bitonic_step<<<g1(PAD), 256, 0, stream>>>(keys, j, kk);
    pool_init<<<g1(N), 256, 0, stream>>>(merged, cluster, nscore, N);
    pool_scan<<<1, 1, 0, stream>>>(keys, ei, E, scv, merged, cluster, nscore, valid);
    fill_f<<<g1((long)N * C), 256, 0, stream>>>(X, 0.0f, (long)N * C);
    pool_agg<<<dim3(N), 256, 0, stream>>>(T1, cluster, X, C);
    pool_scale_relu<<<g1((long)N * C), 256, 0, stream>>>(X, nscore, N, C);
    remap_edges<<<g1(2 * E), 256, 0, stream>>>(ei, cluster, eo, 2 * E);
  };

  // ----- graph branch -----
  embed_gather<<<g1((long)N * 20 * ED), 256, 0, stream>>>(graph_x, embed1w, X, (long)N * 20, ED);
  fill_i<<<g1(N), 256, 0, stream>>>(valid, 1, N);
  run_gcn(2000, 2000, gcnW[0], gcnB[0], edge_idx);
  run_pool(2000, poolW[0], poolB[0], edge_idx, eiA);
  run_gcn(2000, 4000, gcnW[1], gcnB[1], eiA);
  run_pool(4000, poolW[1], poolB[1], eiA, eiB);
  run_gcn(4000, 4000, gcnW[2], gcnB[2], eiB);
  run_pool(4000, poolW[2], poolB[2], eiB, eiA);
  batch_bounds<<<1, 32, 0, stream>>>(batch, N, NB, startb);
  gmax<<<dim3((4000 + 255) / 256, NB), 256, 0, stream>>>(X, valid, startb, 4000, xg);

  // ----- token / GRU branch -----
  embed_gather<<<g1((long)NB * T * ED), 256, 0, stream>>>(tokens, embedw, tokemb, (long)NB * T, ED);
  const float* gin[3]  = {tokemb, Y0, Y1};
  float*       gout[3] = {Y0, Y1, Y0};
  const int din[3] = {ED, 2 * H, 2 * H};
  for (int l = 0; l < 3; ++l) {
    const float* wih = (const float*)d_in[18 + 4 * l];
    const float* whh = (const float*)d_in[19 + 4 * l];
    const float* bih = (const float*)d_in[20 + 4 * l];
    const float* bhh = (const float*)d_in[21 + 4 * l];
    // GI = X @ W_ih^T + b_ih ; W_ih is [600, din] == [N,K] layout already
    conv(gin[l], A16, (long)NB * T * din[l]);
    conv(wih, Wt, (long)3 * H * din[l]);
    gemm16(A16, Wt, bih, GIf, NB * T, 3 * H, din[l], din[l], din[l], 3 * H, 0, stream);
    conv(wih + (long)3 * H * din[l], Wt, (long)3 * H * din[l]);
    gemm16(A16, Wt, bih + 3 * H, GIb, NB * T, 3 * H, din[l], din[l], din[l], 3 * H, 0, stream);
    gru_rec<<<2, 1024, 0, stream>>>(GIf, GIb, whh, bhh, gout[l], x1, l, T);
  }

  // ----- head -----
  concat_z<<<g1(16 * 5200), 256, 0, stream>>>(xg, x1, z);
  conv(z, A16, (long)NB * 5200);
  convT(lin1w, Wt, 5200, 1000);
  gemm16(A16, Wt, lin1b, z1, NB, 1000, 5200, 5200, 5200, 1000, 1, stream);
  conv(z1, A16, (long)NB * 1000);
  convT(lin2w, Wt, 1000, 500);
  gemm16(A16, Wt, lin2b, z2, NB, 500, 1000, 1000, 1000, 500, 1, stream);
  conv(z2, A16, (long)NB * 500);
  convT(lin3w, Wt, 500, 4);
  gemm16(A16, Wt, lin3b, (float*)d_out, NB, 4, 500, 500, 500, 4, 1, stream);
}
